// StateFeatureSelfAttention_57483842290266
// MI455X (gfx1250) — matrix operand order repaired and checked by reading, not compiled, not run on hardware
//
#include <hip/hip_runtime.h>
#include <hip/hip_bf16.h>

typedef __bf16 bf16_t;
typedef __attribute__((ext_vector_type(16))) __bf16 v16bf;
typedef __attribute__((ext_vector_type(8)))  __bf16 v8bf;
typedef __attribute__((ext_vector_type(2)))  __bf16 v2bf;
typedef __attribute__((ext_vector_type(8)))  float  v8f;
typedef __attribute__((ext_vector_type(4)))  float  v4f;

constexpr int cE   = 1024;   // embed dim
constexpr int cH   = 16;     // heads
constexpr int cD   = 64;     // head dim
constexpr int cHID = 1024;   // hidden dim
constexpr int cN   = 8192;   // B*S tokens
constexpr int cSIX = 6;      // states per token
constexpr float cEPS = 1e-5f;

// ---------------- GEMM config ----------------
constexpr int BM  = 128;
constexpr int BN  = 128;
constexpr int BK  = 64;
constexpr int LDT = 72;      // padded LDS row stride (bf16 elems) -> conflict-free b128 reads
constexpr int KTILES = cE / BK;  // 16

union FragU { v16bf v; v8bf h[2]; };

__device__ __forceinline__ void storeC(float* p, float v)  { *p = v; }
__device__ __forceinline__ void storeC(bf16_t* p, float v) { *p = (bf16_t)v; }

// async 16B global -> LDS copy (gfx1250, tracked by ASYNCcnt)
__device__ __forceinline__ void async_cp16(unsigned lds_off, const void* gptr) {
  asm volatile("global_load_async_to_lds_b128 %0, %1, off"
               :: "v"(lds_off), "v"(gptr) : "memory");
}
__device__ __forceinline__ void async_wait0() {
  asm volatile("s_wait_asynccnt 0x0" ::: "memory");
}

// C[M,1024] = A[M,1024](bf16) @ W[1024,1024]^T(bf16) + bias
// Double-buffered LDS tiles filled with global_load_async_to_lds_b128.
template <typename OT, bool LEAKY>
__global__ __launch_bounds__(256) void gemm_bf16_wmma(
    const bf16_t* __restrict__ A, const bf16_t* __restrict__ W,
    const float* __restrict__ bias, OT* __restrict__ C)
{
  __shared__ bf16_t At[2][BM * LDT];
  __shared__ bf16_t Bt[2][BN * LDT];

  const int tid  = threadIdx.x;
  const int lane = tid & 31;
  const int wid  = tid >> 5;
  const int waveM = (wid >> 2) * 64;   // 2 wave-rows
  const int waveN = (wid & 3) * 32;    // 4 wave-cols
  const int lm = lane & 15;
  const int hi = lane >> 4;            // lane group 0/1
  const int rowBase = blockIdx.x * BM;
  const int colBase = blockIdx.y * BN;

  const int srow = tid >> 3;           // 0..31 (rows per staging pass)
  const int skc  = (tid & 7) * 8;      // k-chunk of 8 bf16 (16B)

  v8f zero = {};
  v8f acc[4][2];
  #pragma unroll
  for (int i = 0; i < 4; ++i)
    #pragma unroll
    for (int j = 0; j < 2; ++j) acc[i][j] = zero;

  auto stage = [&](int buf, int k0) {
    #pragma unroll
    for (int p = 0; p < 4; ++p) {
      const int r = p * 32 + srow;
      const bf16_t* ga = A + (size_t)(rowBase + r) * cE + k0 + skc;
      const bf16_t* gb = W + (size_t)(colBase + r) * cE + k0 + skc;
      const unsigned la = (unsigned)(size_t)&At[buf][r * LDT + skc];
      const unsigned lb = (unsigned)(size_t)&Bt[buf][r * LDT + skc];
      async_cp16(la, ga);
      async_cp16(lb, gb);
    }
  };

  stage(0, 0);
  async_wait0();
  __syncthreads();

  for (int kt = 0; kt < KTILES; ++kt) {
    const int cur = kt & 1;
    if (kt + 1 < KTILES) stage(cur ^ 1, (kt + 1) * BK);  // prefetch next tile (async)

    #pragma unroll
    for (int kk = 0; kk < BK; kk += 32) {
      FragU af[4]; FragU bfr[2];
      const int akb = kk + (hi ? 8 : 0);       // A frag: K runs {akb..+7, akb+16..+23}
      #pragma unroll
      for (int i = 0; i < 4; ++i) {
        const bf16_t* ap = &At[cur][(waveM + 16 * i + lm) * LDT];
        af[i].h[0] = *(const v8bf*)(ap + akb);
        af[i].h[1] = *(const v8bf*)(ap + akb + 16);
      }
      const int bkb = kk + (hi ? 8 : 0);       // B frag: K runs {bkb..+7, bkb+16..+23}
      #pragma unroll
      for (int j = 0; j < 2; ++j) {
        const bf16_t* bp = &Bt[cur][(waveN + 16 * j + lm) * LDT];
        bfr[j].h[0] = *(const v8bf*)(bp + bkb);
        bfr[j].h[1] = *(const v8bf*)(bp + bkb + 16);
      }
      #pragma unroll
      for (int i = 0; i < 4; ++i)
        #pragma unroll
        for (int j = 0; j < 2; ++j)
          acc[i][j] = __builtin_amdgcn_wmma_f32_16x16x32_bf16(
              false, af[i].v, false, bfr[j].v, (short)0, acc[i][j], false, false);
    }

    if (kt + 1 < KTILES) {
      async_wait0();       // our async copies into the other buffer have landed
      __syncthreads();     // everyone done reading cur + done staging next
    }
  }

  // epilogue: bias (+ optional LeakyReLU), store f32 or bf16
  #pragma unroll
  for (int j = 0; j < 2; ++j) {
    const int col = colBase + waveN + 16 * j + lm;
    const float bv = bias[col];
    #pragma unroll
    for (int i = 0; i < 4; ++i) {
      #pragma unroll
      for (int vg = 0; vg < 8; ++vg) {
        const int row = rowBase + waveM + 16 * i + vg + 8 * hi;
        float v = acc[i][j][vg] + bv;
        if (LEAKY) v = (v >= 0.0f) ? v : 0.01f * v;
        storeC(&C[(size_t)row * cE + col], v);
      }
    }
  }
}

// ---------------- tiny 6x6 attention, one wave per (token, head) ----------------
__device__ __forceinline__ float wave_sum32(float x) {
  #pragma unroll
  for (int m = 16; m >= 1; m >>= 1) x += __shfl_xor(x, m, 32);
  return x;
}

__global__ __launch_bounds__(256) void attn6_kernel(
    bf16_t* __restrict__ Q, const bf16_t* __restrict__ K, const bf16_t* __restrict__ V)
{
  const int gtid = blockIdx.x * blockDim.x + threadIdx.x;
  const int widx = gtid >> 5;
  const int lane = threadIdx.x & 31;
  const int n = widx / cH;
  const int h = widx % cH;
  const int d0 = lane * 2;

  float qf[cSIX][2], kf[cSIX][2], vf[cSIX][2];
  #pragma unroll
  for (int j = 0; j < cSIX; ++j) {
    const size_t base = ((size_t)j * cN + n) * cE + h * cD + d0;
    v2bf qv = *(const v2bf*)&Q[base];
    v2bf kv = *(const v2bf*)&K[base];
    v2bf vv = *(const v2bf*)&V[base];
    qf[j][0] = (float)qv.x; qf[j][1] = (float)qv.y;
    kf[j][0] = (float)kv.x; kf[j][1] = (float)kv.y;
    vf[j][0] = (float)vv.x; vf[j][1] = (float)vv.y;
  }

  float sc[cSIX][cSIX];
  #pragma unroll
  for (int a = 0; a < cSIX; ++a)
    #pragma unroll
    for (int b = 0; b < cSIX; ++b) {
      float p = qf[a][0] * kf[b][0] + qf[a][1] * kf[b][1];
      sc[a][b] = wave_sum32(p) * 0.125f;   // 1/sqrt(64)
    }

  #pragma unroll
  for (int a = 0; a < cSIX; ++a) {
    float mx = sc[a][0];
    #pragma unroll
    for (int b = 1; b < cSIX; ++b) mx = fmaxf(mx, sc[a][b]);
    float s = 0.0f;
    #pragma unroll
    for (int b = 0; b < cSIX; ++b) { sc[a][b] = __expf(sc[a][b] - mx); s += sc[a][b]; }
    const float inv = 1.0f / s;
    #pragma unroll
    for (int b = 0; b < cSIX; ++b) sc[a][b] *= inv;
  }

  #pragma unroll
  for (int a = 0; a < cSIX; ++a) {
    float c0 = 0.0f, c1 = 0.0f;
    #pragma unroll
    for (int b = 0; b < cSIX; ++b) { c0 += sc[a][b] * vf[b][0]; c1 += sc[a][b] * vf[b][1]; }
    const size_t base = ((size_t)a * cN + n) * cE + h * cD + d0;
    v2bf cv; cv.x = (bf16_t)c0; cv.y = (bf16_t)c1;
    *(v2bf*)&Q[base] = cv;                 // ctx overwrites Q in place
  }
}

// ---------------- LN helpers ----------------
__device__ __forceinline__ float block_sum256(float x, float* sred) {
  x = wave_sum32(x);
  const int wid = threadIdx.x >> 5;
  if ((threadIdx.x & 31) == 0) sred[wid] = x;
  __syncthreads();
  float t = 0.0f;
  #pragma unroll
  for (int i = 0; i < 8; ++i) t += sred[i];
  __syncthreads();
  return t;
}

// residual + LN1 + softmax-weighted fusion over 6 states -> fused (bf16)
__global__ __launch_bounds__(256) void ln1_fusion_kernel(
    const float* __restrict__ Out, const float* __restrict__ SF,
    const float* __restrict__ g, const float* __restrict__ b,
    const float* __restrict__ fw, bf16_t* __restrict__ Fused)
{
  __shared__ float sred[8];
  const int n = blockIdx.x;
  const int e0 = threadIdx.x * 4;

  float w[cSIX];
  float wm = -1e30f;
  #pragma unroll
  for (int j = 0; j < cSIX; ++j) { w[j] = fw[j]; wm = fmaxf(wm, w[j]); }
  float wsum = 0.0f;
  #pragma unroll
  for (int j = 0; j < cSIX; ++j) { w[j] = __expf(w[j] - wm); wsum += w[j]; }
  const float winv = 1.0f / wsum;
  #pragma unroll
  for (int j = 0; j < cSIX; ++j) w[j] *= winv;

  const v4f gg = *(const v4f*)&g[e0];
  const v4f bb = *(const v4f*)&b[e0];
  float facc[4] = {0.0f, 0.0f, 0.0f, 0.0f};

  for (int j = 0; j < cSIX; ++j) {
    const size_t base = ((size_t)j * cN + n) * cE + e0;
    v4f o = *(const v4f*)&Out[base];
    v4f s = *(const v4f*)&SF[base];
    float val[4] = { o.x + s.x, o.y + s.y, o.z + s.z, o.w + s.w };
    float ls = val[0] + val[1] + val[2] + val[3];
    float lq = val[0]*val[0] + val[1]*val[1] + val[2]*val[2] + val[3]*val[3];
    const float sum = block_sum256(ls, sred);
    const float sq  = block_sum256(lq, sred);
    const float mu  = sum * (1.0f / cE);
    const float var = sq * (1.0f / cE) - mu * mu;
    const float rs  = rsqrtf(var + cEPS);
    #pragma unroll
    for (int i = 0; i < 4; ++i) {
      const float nv = (val[i] - mu) * rs * gg[i] + bb[i];
      facc[i] += w[j] * nv;
    }
  }
  bf16_t* dst = &Fused[(size_t)n * cE + e0];
  dst[0] = (bf16_t)facc[0]; dst[1] = (bf16_t)facc[1];
  dst[2] = (bf16_t)facc[2]; dst[3] = (bf16_t)facc[3];
}

// LN2 over hidden (LeakyReLU already applied in GEMM epilogue) -> f32 output
__global__ __launch_bounds__(256) void ln2_kernel(
    const float* __restrict__ Hdn, const float* __restrict__ g,
    const float* __restrict__ b, float* __restrict__ Outp)
{
  __shared__ float sred[8];
  const int n = blockIdx.x;
  const int e0 = threadIdx.x * 4;
  v4f hv = *(const v4f*)&Hdn[(size_t)n * cHID + e0];
  float val[4] = { hv.x, hv.y, hv.z, hv.w };
  float ls = val[0] + val[1] + val[2] + val[3];
  float lq = val[0]*val[0] + val[1]*val[1] + val[2]*val[2] + val[3]*val[3];
  const float sum = block_sum256(ls, sred);
  const float sq  = block_sum256(lq, sred);
  const float mu  = sum * (1.0f / cHID);
  const float var = sq * (1.0f / cHID) - mu * mu;
  const float rs  = rsqrtf(var + cEPS);
  const v4f gg = *(const v4f*)&g[e0];
  const v4f bb = *(const v4f*)&b[e0];
  v4f o;
  o.x = (val[0] - mu) * rs * gg[0] + bb[0];
  o.y = (val[1] - mu) * rs * gg[1] + bb[1];
  o.z = (val[2] - mu) * rs * gg[2] + bb[2];
  o.w = (val[3] - mu) * rs * gg[3] + bb[3];
  *(v4f*)&Outp[(size_t)n * cHID + e0] = o;
}

// fp32 -> bf16 conversion (weights + activations)
__global__ __launch_bounds__(256) void cvt_kernel(
    const float* __restrict__ src, bf16_t* __restrict__ dst, int n4)
{
  const int i = blockIdx.x * blockDim.x + threadIdx.x;
  if (i < n4) {
    v4f f = *(const v4f*)&src[(size_t)i * 4];
    bf16_t* d = &dst[(size_t)i * 4];
    d[0] = (bf16_t)f.x; d[1] = (bf16_t)f.y; d[2] = (bf16_t)f.z; d[3] = (bf16_t)f.w;
  }
}

extern "C" void kernel_launch(void* const* d_in, const int* in_sizes, int n_in,
                              void* d_out, int out_size, void* d_ws, size_t ws_size,
                              hipStream_t stream) {
  (void)in_sizes; (void)n_in; (void)out_size; (void)ws_size;
  const float* sf  = (const float*)d_in[0];   // (6,B,S,E) == rows r = j*N + n
  const float* Wq  = (const float*)d_in[1];
  const float* bq  = (const float*)d_in[2];
  const float* Wk  = (const float*)d_in[3];
  const float* bk  = (const float*)d_in[4];
  const float* Wv  = (const float*)d_in[5];
  const float* bv  = (const float*)d_in[6];
  const float* Wo  = (const float*)d_in[7];
  const float* bo  = (const float*)d_in[8];
  const float* g1  = (const float*)d_in[9];
  const float* b1  = (const float*)d_in[10];
  const float* fw  = (const float*)d_in[11];
  const float* Wf  = (const float*)d_in[12];
  const float* bfv = (const float*)d_in[13];
  const float* g2  = (const float*)d_in[14];
  const float* b2  = (const float*)d_in[15];
  float* outp = (float*)d_out;

  const size_t MB = 1ull << 20;
  char* ws = (char*)d_ws;
  bf16_t* Xb   = (bf16_t*)(ws + 0);         // 96MB activation bf16 (dead after QKV)
  bf16_t* Qb   = (bf16_t*)(ws + 96 * MB);   // 96MB; ctx overwrites in place
  bf16_t* Kb   = (bf16_t*)(ws + 192 * MB);  // 96MB
  bf16_t* Vb   = (bf16_t*)(ws + 288 * MB);  // 96MB
  float*  OutM = (float*)(ws + 192 * MB);   // 192MB, reuses K/V space after attention
  bf16_t* Fus  = (bf16_t*)(ws + 0);         // 16MB, reuses Xb space
  float*  Hdn  = (float*)(ws + 16 * MB);    // 32MB, reuses Xb space
  bf16_t* Wqb  = (bf16_t*)(ws + 384 * MB);  // 2MB each
  bf16_t* Wkb  = (bf16_t*)(ws + 386 * MB);
  bf16_t* Wvb  = (bf16_t*)(ws + 388 * MB);
  bf16_t* Wob  = (bf16_t*)(ws + 390 * MB);
  bf16_t* Wfb  = (bf16_t*)(ws + 392 * MB);

  // 1) weights + activations fp32 -> bf16
  const int n4w = (cE * cE) / 4;
  cvt_kernel<<<n4w / 256, 256, 0, stream>>>(Wq, Wqb, n4w);
  cvt_kernel<<<n4w / 256, 256, 0, stream>>>(Wk, Wkb, n4w);
  cvt_kernel<<<n4w / 256, 256, 0, stream>>>(Wv, Wvb, n4w);
  cvt_kernel<<<n4w / 256, 256, 0, stream>>>(Wo, Wob, n4w);
  cvt_kernel<<<n4w / 256, 256, 0, stream>>>(Wf, Wfb, n4w);
  const int n4x = (cSIX * cN * cE) / 4;
  cvt_kernel<<<n4x / 256, 256, 0, stream>>>(sf, Xb, n4x);

  // 2) Q/K/V projections (all-bf16, async-LDS staged, double buffered)
  dim3 gB(256);
  dim3 gG((cSIX * cN) / BM, cE / BN);   // (384, 8)
  gemm_bf16_wmma<bf16_t, false><<<gG, gB, 0, stream>>>(Xb, Wqb, bq, Qb);
  gemm_bf16_wmma<bf16_t, false><<<gG, gB, 0, stream>>>(Xb, Wkb, bk, Kb);
  gemm_bf16_wmma<bf16_t, false><<<gG, gB, 0, stream>>>(Xb, Wvb, bv, Vb);

  // 3) tiny 6x6 attention; ctx written over Q
  attn6_kernel<<<(cN * cH) / 8, 256, 0, stream>>>(Qb, Kb, Vb);

  // 4) output projection: bf16 ctx -> f32 out
  gemm_bf16_wmma<float, false><<<gG, gB, 0, stream>>>(Qb, Wob, bo, OutM);

  // 5) residual + LN1 + weighted fusion -> bf16 fused
  ln1_fusion_kernel<<<cN, 256, 0, stream>>>(OutM, sf, g1, b1, fw, Fus);

  // 6) FFN with LeakyReLU epilogue -> f32 hidden
  dim3 g2G(cN / BM, cHID / BN);         // (64, 8)
  gemm_bf16_wmma<float, true><<<g2G, gB, 0, stream>>>(Fus, Wfb, bfv, Hdn);

  // 7) LN2 -> final output
  ln2_kernel<<<cN, 256, 0, stream>>>(Hdn, g2, b2, outp);
}
